// DeformNet_42932493091277
// MI455X (gfx1250) — matrix operand order repaired and checked by reading, not compiled, not run on hardware
//
#include <hip/hip_runtime.h>
#include <hip/hip_bf16.h>

typedef _Float16 v8h  __attribute__((ext_vector_type(8)));
typedef _Float16 v16h __attribute__((ext_vector_type(16)));
typedef float    v8f  __attribute__((ext_vector_type(8)));

#define BN_EPS 1e-5f

// Geometry
#define B_    2
#define H2    112
#define W2D   112
#define HP    114          // padded 112+2
#define NPOS  (B_*H2*W2D)  // 25088 spatial positions
#define MTILES (NPOS/16)   // 1568

// LDS staging for deform A tile: 16 rows x 256 c, row padded to 264 halfs
#define AROW 264

// Workspace layout (bytes, every region 512B aligned)
static constexpr size_t OFF_T1     = 0;                       // conv1+relu, NHWC f32 (2,224,224,64)
static constexpr size_t OFF_STATS1 = OFF_T1     + 25690112;   // [mean(64), inv_std(64)]
static constexpr size_t OFF_T2P    = OFF_STATS1 + 512;        // pooled+BN1, NHWC f16 padded (2,114,114,64)
static constexpr size_t OFF_BT2    = OFF_T2P    + 3326976;    // w2 panel f16 [256][576]
static constexpr size_t OFF_T3     = OFF_BT2    + 294912;     // relu(conv2), NHWC f32 (2,112,112,256)
static constexpr size_t OFF_STATS2 = OFF_T3     + 25690112;   // [mean(256), inv_std(256)]
static constexpr size_t OFF_HP     = OFF_STATS2 + 2048;       // h (BN2 out), NHWC f16 padded (2,114,114,256)
static constexpr size_t OFF_BTO    = OFF_HP     + 13307904;   // w_off panel f16 [32][2304] (rows 18..31 zero)
static constexpr size_t OFF_OFFB   = OFF_BTO    + 147456;     // offset maps NHWC f32 (2,112,112,18)
static constexpr size_t OFF_DESCI  = OFF_OFFB   + 1806336;    // int4  per (pos,tap): 4 corner offsets
static constexpr size_t OFF_DESCW  = OFF_DESCI  + 3612672;    // float4 per (pos,tap): 4 bilinear weights
static constexpr size_t OFF_BT4    = OFF_DESCW  + 3612672;    // w4 panel f16 [256][2304]

__device__ inline v16h cat16(v8h lo, v8h hi) {
  return __builtin_shufflevector(lo, hi, 0,1,2,3,4,5,6,7,8,9,10,11,12,13,14,15);
}

// ---------------- Stage 1: conv 3->64 + bias + relu (f32 direct, tiny K) ----------------
__global__ __launch_bounds__(256) void k_conv1(const float* __restrict__ x,
                                               const float* __restrict__ w1,
                                               const float* __restrict__ b1,
                                               float* __restrict__ t1) {
  int t = blockIdx.x * 256 + threadIdx.x;            // 2*224*224*64 threads
  int co = t & 63;
  int pos = t >> 6;
  int xx = pos % 224; int r = pos / 224; int yy = r % 224; int b = r / 224;
  float acc = b1[co];
#pragma unroll
  for (int ci = 0; ci < 3; ++ci)
#pragma unroll
    for (int kh = 0; kh < 3; ++kh) {
      int iy = yy + kh - 1;
      if ((unsigned)iy >= 224u) continue;
#pragma unroll
      for (int kw = 0; kw < 3; ++kw) {
        int ix = xx + kw - 1;
        if ((unsigned)ix >= 224u) continue;
        acc += x[((b * 3 + ci) * 224 + iy) * 224 + ix] * w1[((co * 3 + ci) * 3 + kh) * 3 + kw];
      }
    }
  t1[(size_t)pos * 64 + co] = fmaxf(acc, 0.f);
}

// ---------------- BN batch statistics: one block per channel ----------------
__global__ __launch_bounds__(256) void k_bnstats(const float* __restrict__ src, int C, int Mtot,
                                                 float* __restrict__ stats) {
  __shared__ float ss[256], sq[256];
  int c = blockIdx.x, t = threadIdx.x;
  float s = 0.f, q = 0.f;
  for (int i = t; i < Mtot; i += 256) {
    float v = src[(size_t)i * C + c];
    s += v; q += v * v;
  }
  ss[t] = s; sq[t] = q; __syncthreads();
  for (int w = 128; w > 0; w >>= 1) {
    if (t < w) { ss[t] += ss[t + w]; sq[t] += sq[t + w]; }
    __syncthreads();
  }
  if (t == 0) {
    float mean = ss[0] / (float)Mtot;
    float var  = sq[0] / (float)Mtot - mean * mean;   // biased, like jnp.var
    stats[c] = mean;
    stats[C + c] = rsqrtf(var + BN_EPS);
  }
}

// ---------------- BN1 apply + avgpool 2x2 + pad -> NHWC f16 (2,114,114,64) ----------------
__global__ __launch_bounds__(256) void k_bn1_pool_pad(const float* __restrict__ t1,
                                                      const float* __restrict__ stats,
                                                      const float* __restrict__ g,
                                                      const float* __restrict__ be,
                                                      _Float16* __restrict__ t2p) {
  int t = blockIdx.x * 256 + threadIdx.x;            // 2*114*114*64 threads
  int c = t & 63; int p = t >> 6;
  int xp = p % 114; int q = p / 114; int yp = q % 114; int b = q / 114;
  float v = 0.f;
  if (xp >= 1 && xp <= 112 && yp >= 1 && yp <= 112) {
    int y = yp - 1, x = xp - 1;
    size_t base = ((size_t)(b * 224 + 2 * y) * 224 + 2 * x) * 64 + c;
    float s = 0.25f * (t1[base] + t1[base + 64] + t1[base + 224 * 64] + t1[base + 224 * 64 + 64]);
    v = (s - stats[c]) * stats[64 + c] * g[c] + be[c];   // affine BN commutes with avgpool
  }
  t2p[t] = (_Float16)v;
}

// ---------------- Weight panel prep (f16, Bt[n][k], k = tap*C + c) ----------------
__global__ __launch_bounds__(256) void k_prep_w2(const float* __restrict__ w2, _Float16* __restrict__ Bt) {
  int t = blockIdx.x * 256 + threadIdx.x;            // 256*576
  int n = t / 576, k = t % 576;
  int tap = k / 64, c = k % 64;
  Bt[t] = (_Float16)w2[(n * 64 + c) * 9 + tap];
}
__global__ __launch_bounds__(256) void k_prep_woff(const float* __restrict__ w_off, _Float16* __restrict__ Bt) {
  int t = blockIdx.x * 256 + threadIdx.x;            // 32*2304, rows >=18 zero-padded
  int n = t / 2304, k = t % 2304;
  int tap = k / 256, c = k % 256;
  Bt[t] = (_Float16)(n < 18 ? w_off[(n * 256 + c) * 9 + tap] : 0.f);
}
__global__ __launch_bounds__(256) void k_prep_w4(const float* __restrict__ w4, _Float16* __restrict__ Bt) {
  int t = blockIdx.x * 256 + threadIdx.x;            // 256*2304
  int n = t / 2304, k = t % 2304;
  int tap = k / 256, c = k % 256;
  Bt[t] = (_Float16)w4[(n * 256 + c) * 9 + tap];
}

// ---------------- conv2 64->256 implicit GEMM via WMMA + bias + relu ----------------
// M=25088, N=256, K=576. One wave = one 16x16 C tile, 4 waves/block.
__global__ __launch_bounds__(128) void k_conv2_wmma(const _Float16* __restrict__ t2p,
                                                    const _Float16* __restrict__ Bt,
                                                    const float* __restrict__ b2,
                                                    float* __restrict__ t3) {
  const int lane = threadIdx.x & 31, wave = threadIdx.x >> 5;
  const int half = lane >> 4, l15 = lane & 15;
  const int mtile = blockIdx.x;
  const int ntile = blockIdx.y * 4 + wave;
  const int mrow = mtile * 16 + l15;
  const int b = mrow / 12544, rem = mrow % 12544;
  const int y = rem / 112, x = rem % 112;
  const int ncol = ntile * 16 + l15;
  const _Float16* brow = Bt + (size_t)ncol * 576;
  v8f acc = {};
  for (int tap = 0; tap < 9; ++tap) {
    const int kh = tap / 3, kw = tap - kh * 3;
    const _Float16* arow = t2p + (((size_t)(b * 114 + y + kh) * 114) + (x + kw)) * 64;
#pragma unroll
    for (int cb = 0; cb < 64; cb += 32) {
      v8h a0 = *(const v8h*)(arow + cb + 8 * half);
      v8h a1 = *(const v8h*)(arow + cb + 16 + 8 * half);
      v16h A = cat16(a0, a1);
      int k0 = tap * 64 + cb;
      v8h bb0 = *(const v8h*)(brow + k0 + 8 * half);
      v8h bb1 = *(const v8h*)(brow + k0 + 16 + 8 * half);
      v16h Bm = cat16(bb0, bb1);
      acc = __builtin_amdgcn_wmma_f32_16x16x32_f16(false, A, false, Bm, (short)0, acc, false, false);
    }
  }
  float bias = b2[ncol];
#pragma unroll
  for (int r = 0; r < 8; ++r) {
    int mg = mtile * 16 + r + 8 * half;
    t3[(size_t)mg * 256 + ncol] = fmaxf(acc[r] + bias, 0.f);
  }
}

// ---------------- BN2 apply + pad -> h in NHWC f16 (2,114,114,256) ----------------
__global__ __launch_bounds__(256) void k_bn2_pad(const float* __restrict__ t3,
                                                 const float* __restrict__ stats,
                                                 const float* __restrict__ g,
                                                 const float* __restrict__ be,
                                                 _Float16* __restrict__ hp) {
  int t = blockIdx.x * 256 + threadIdx.x;            // 2*114*114*256 threads
  int c = t & 255; int p = t >> 8;
  int xp = p % 114; int q = p / 114; int yp = q % 114; int b = q / 114;
  float v = 0.f;
  if (xp >= 1 && xp <= 112 && yp >= 1 && yp <= 112) {
    int m = (b * 112 + (yp - 1)) * 112 + (xp - 1);
    v = (t3[(size_t)m * 256 + c] - stats[c]) * stats[256 + c] * g[c] + be[c];
  }
  hp[t] = (_Float16)v;
}

// ---------------- offset conv 256->18 (N padded to 32) via WMMA ----------------
__global__ __launch_bounds__(64) void k_convoff_wmma(const _Float16* __restrict__ hp,
                                                     const _Float16* __restrict__ Bt,
                                                     const float* __restrict__ b_off,
                                                     float* __restrict__ offb) {
  const int lane = threadIdx.x & 31, wave = threadIdx.x >> 5;   // 2 waves
  const int half = lane >> 4, l15 = lane & 15;
  const int mtile = blockIdx.x;
  const int mrow = mtile * 16 + l15;
  const int b = mrow / 12544, rem = mrow % 12544;
  const int y = rem / 112, x = rem % 112;
  const int ncol = wave * 16 + l15;                  // 0..31 (only <18 stored)
  const _Float16* brow = Bt + (size_t)ncol * 2304;
  v8f acc = {};
  for (int tap = 0; tap < 9; ++tap) {
    const int kh = tap / 3, kw = tap - kh * 3;
    const _Float16* arow = hp + (((size_t)(b * 114 + y + kh) * 114) + (x + kw)) * 256;
    for (int cb = 0; cb < 256; cb += 32) {
      v8h a0 = *(const v8h*)(arow + cb + 8 * half);
      v8h a1 = *(const v8h*)(arow + cb + 16 + 8 * half);
      v16h A = cat16(a0, a1);
      int k0 = tap * 256 + cb;
      v8h bb0 = *(const v8h*)(brow + k0 + 8 * half);
      v8h bb1 = *(const v8h*)(brow + k0 + 16 + 8 * half);
      v16h Bm = cat16(bb0, bb1);
      acc = __builtin_amdgcn_wmma_f32_16x16x32_f16(false, A, false, Bm, (short)0, acc, false, false);
    }
  }
  float bias = (ncol < 18) ? b_off[ncol] : 0.f;
#pragma unroll
  for (int r = 0; r < 8; ++r) {
    int mg = mtile * 16 + r + 8 * half;
    if (ncol < 18) offb[(size_t)mg * 18 + ncol] = acc[r] + bias;
  }
}

// ---------------- bilinear sampling descriptors (reference semantics) ----------------
__global__ __launch_bounds__(256) void k_sample_prep(const float* __restrict__ offb,
                                                     int4* __restrict__ descI,
                                                     float4* __restrict__ descW) {
  int t = blockIdx.x * 256 + threadIdx.x;            // NPOS*9 threads
  if (t >= NPOS * 9) return;
  int tap = t % 9; int m = t / 9;
  int x = m % 112; int r = m / 112; int y = r % 112; int b = r / 112;
  int kh = tap / 3, kw = tap - kh * 3;
  float ox = offb[(size_t)m * 18 + 2 * tap];
  float oy = offb[(size_t)m * 18 + 2 * tap + 1];
  float px = (float)(y + kh) + ox;                   // padded coords: 1 + (y) + (kh-1) + off
  float py = (float)(x + kw) + oy;
  float fpx = floorf(px), fpy = floorf(py);
  int ltx = (int)fminf(fmaxf(fpx,       0.f), 113.f);
  int lty = (int)fminf(fmaxf(fpy,       0.f), 113.f);
  int rbx = (int)fminf(fmaxf(fpx + 1.f, 0.f), 113.f);
  int rby = (int)fminf(fmaxf(fpy + 1.f, 0.f), 113.f);
  bool mx = (px < 1.f) || (px > 112.f);
  bool my = (py < 1.f) || (py > 112.f);
  px = fminf(fmaxf(mx ? fpx : px, 0.f), 113.f);
  py = fminf(fmaxf(my ? fpy : py, 0.f), 113.f);
  float dxl = (float)ltx - px, dyl = (float)lty - py;
  float dxr = (float)rbx - px, dyr = (float)rby - py;
  float g_lt = (1.f + dxl) * (1.f + dyl);
  float g_rb = (1.f - dxr) * (1.f - dyr);
  float g_lb = (1.f + dxl) * (1.f - dyr);
  float g_rt = (1.f - dxr) * (1.f + dyl);
  int base = b * 114 * 114 * 256;
  descI[t] = make_int4(base + (ltx * 114 + lty) * 256,
                       base + (rbx * 114 + rby) * 256,
                       base + (ltx * 114 + rby) * 256,
                       base + (rbx * 114 + lty) * 256);
  descW[t] = make_float4(g_lt, g_rb, g_lb, g_rt);
}

__device__ inline v8h sample8(const _Float16* __restrict__ hp, const int4 oi, const float4 w, int off) {
  const v8h p0 = *(const v8h*)(hp + oi.x + off);
  const v8h p1 = *(const v8h*)(hp + oi.y + off);
  const v8h p2 = *(const v8h*)(hp + oi.z + off);
  const v8h p3 = *(const v8h*)(hp + oi.w + off);
  v8h rr;
#pragma unroll
  for (int j = 0; j < 8; ++j)
    rr[j] = (_Float16)(w.x * (float)p0[j] + w.y * (float)p1[j] + w.z * (float)p2[j] + w.w * (float)p3[j]);
  return rr;
}

// ---------------- deformable contraction: LDS-staged gather-GEMM via WMMA -> d_out NCHW f32 ----
// Block = 128 threads (4 waves) covering M=16 x N=128; A tile (16x256 per tap) is sampled
// cooperatively ONCE into LDS, then consumed by all 4 waves (2 n-tiles each).
__global__ __launch_bounds__(128) void k_deform_wmma(const _Float16* __restrict__ hp,
                                                     const _Float16* __restrict__ Bt,
                                                     const int4* __restrict__ descI,
                                                     const float4* __restrict__ descW,
                                                     float* __restrict__ out) {
  __shared__ __align__(16) _Float16 smem[16 * AROW];   // 8448 B, row-padded vs bank conflicts
  const int tid  = threadIdx.x;
  const int lane = tid & 31, wave = tid >> 5;
  const int half = lane >> 4, l15 = lane & 15;
  const int mtile = blockIdx.x;
  const int nbase = blockIdx.y * 128;
  const int ncol0 = nbase + wave * 32 + l15;
  const int ncol1 = ncol0 + 16;
  const _Float16* brow0 = Bt + (size_t)ncol0 * 2304;
  const _Float16* brow1 = Bt + (size_t)ncol1 * 2304;
  // cooperative sampling role: 128 threads = 16 rows x 8 segments of 32 channels
  const int srow = tid >> 3, sseg = tid & 7;
  const int sd_base = (mtile * 16 + srow) * 9;

  v8f acc0 = {}, acc1 = {};
  for (int tap = 0; tap < 9; ++tap) {
    // ---- build sampled A tile for this tap in LDS (once per block) ----
    {
      const int4   oi = descI[sd_base + tap];
      const float4 w  = descW[sd_base + tap];
      const int c0 = sseg * 32;
#pragma unroll
      for (int j = 0; j < 4; ++j) {
        v8h s = sample8(hp, oi, w, c0 + j * 8);
        *(v8h*)(&smem[srow * AROW + c0 + j * 8]) = s;
      }
    }
    __syncthreads();
    // prefetch next K span of the streamed B panels into cache
    __builtin_prefetch(brow0 + (tap + 1) * 256, 0, 0);
    __builtin_prefetch(brow1 + (tap + 1) * 256, 0, 0);
    // ---- consume: 8 k-steps of 32, two 16x16 WMMA per step per wave ----
    const _Float16* arow = &smem[l15 * AROW];
#pragma unroll
    for (int cb = 0; cb < 256; cb += 32) {
      v8h a0 = *(const v8h*)(arow + cb + 8 * half);
      v8h a1 = *(const v8h*)(arow + cb + 16 + 8 * half);
      v16h A = cat16(a0, a1);
      int k0 = tap * 256 + cb;
      v8h b00 = *(const v8h*)(brow0 + k0 + 8 * half);
      v8h b01 = *(const v8h*)(brow0 + k0 + 16 + 8 * half);
      v16h Bm0 = cat16(b00, b01);
      acc0 = __builtin_amdgcn_wmma_f32_16x16x32_f16(false, A, false, Bm0, (short)0, acc0, false, false);
      v8h b10 = *(const v8h*)(brow1 + k0 + 8 * half);
      v8h b11 = *(const v8h*)(brow1 + k0 + 16 + 8 * half);
      v16h Bm1 = cat16(b10, b11);
      acc1 = __builtin_amdgcn_wmma_f32_16x16x32_f16(false, A, false, Bm1, (short)0, acc1, false, false);
    }
    __syncthreads();
  }
#pragma unroll
  for (int r = 0; r < 8; ++r) {
    int mg = mtile * 16 + r + 8 * half;
    int b = mg / 12544, rem = mg % 12544;
    int y = rem / 112, x = rem % 112;
    out[(((size_t)b * 256 + ncol0) * 112 + y) * 112 + x] = acc0[r];
    out[(((size_t)b * 256 + ncol1) * 112 + y) * 112 + x] = acc1[r];
  }
}

extern "C" void kernel_launch(void* const* d_in, const int* in_sizes, int n_in,
                              void* d_out, int out_size, void* d_ws, size_t ws_size,
                              hipStream_t stream) {
  (void)in_sizes; (void)n_in; (void)out_size; (void)ws_size;
  const float* x     = (const float*)d_in[0];
  const float* w1    = (const float*)d_in[1];
  const float* b1    = (const float*)d_in[2];
  const float* g1    = (const float*)d_in[3];
  const float* be1   = (const float*)d_in[4];
  const float* w2    = (const float*)d_in[5];
  const float* b2    = (const float*)d_in[6];
  const float* g2    = (const float*)d_in[7];
  const float* be2   = (const float*)d_in[8];
  const float* w_off = (const float*)d_in[9];
  const float* b_off = (const float*)d_in[10];
  const float* w4    = (const float*)d_in[11];

  char* ws = (char*)d_ws;
  float*    t1     = (float*)   (ws + OFF_T1);
  float*    stats1 = (float*)   (ws + OFF_STATS1);
  _Float16* t2p    = (_Float16*)(ws + OFF_T2P);
  _Float16* Bt2    = (_Float16*)(ws + OFF_BT2);
  float*    t3     = (float*)   (ws + OFF_T3);
  float*    stats2 = (float*)   (ws + OFF_STATS2);
  _Float16* hp     = (_Float16*)(ws + OFF_HP);
  _Float16* BtO    = (_Float16*)(ws + OFF_BTO);
  float*    offb   = (float*)   (ws + OFF_OFFB);
  int4*     descI  = (int4*)    (ws + OFF_DESCI);
  float4*   descW  = (float4*)  (ws + OFF_DESCW);
  _Float16* Bt4    = (_Float16*)(ws + OFF_BT4);
  float*    out    = (float*)d_out;

  // Weight panels (independent of activations)
  k_prep_w2  <<<576,  256, 0, stream>>>(w2,    Bt2);
  k_prep_woff<<<288,  256, 0, stream>>>(w_off, BtO);
  k_prep_w4  <<<2304, 256, 0, stream>>>(w4,    Bt4);

  // Stage 1
  k_conv1      <<<25088, 256, 0, stream>>>(x, w1, b1, t1);
  k_bnstats    <<<64,    256, 0, stream>>>(t1, 64, 2 * 224 * 224, stats1);
  k_bn1_pool_pad<<<6498, 256, 0, stream>>>(t1, stats1, g1, be1, t2p);

  // Stage 2 (WMMA)
  k_conv2_wmma <<<dim3(MTILES, 4), 128, 0, stream>>>(t2p, Bt2, b2, t3);
  k_bnstats    <<<256,   256, 0, stream>>>(t3, 256, NPOS, stats2);
  k_bn2_pad    <<<25992, 256, 0, stream>>>(t3, stats2, g2, be2, hp);

  // Stage 3 (WMMA, N padded 18->32)
  k_convoff_wmma<<<MTILES, 64, 0, stream>>>(hp, BtO, b_off, offb);

  // Stage 4: descriptors + LDS-staged fused gather-GEMM (WMMA)
  k_sample_prep<<<882, 256, 0, stream>>>(offb, descI, descW);
  k_deform_wmma<<<dim3(MTILES, 2), 128, 0, stream>>>(hp, Bt4, descI, descW, out);
}
